// TransformerBlock_57698590654737
// MI455X (gfx1250) — compile-verified, matrix operand order checked by reading, not run on hardware
//
#include <hip/hip_runtime.h>
#include <hip/hip_bf16.h>

typedef _Float16 v8h  __attribute__((ext_vector_type(8)));
typedef _Float16 v16h __attribute__((ext_vector_type(16)));
typedef float    v8f  __attribute__((ext_vector_type(8)));

#define BB 4
#define NN 4096
#define DD 256
#define HH 128
#define BN (BB*NN)   /* 16384 rows */

// ---------------------------------------------------------------------------
// WMMA fragment load: A-matrix 16x32 f16 layout (ISA 7.12.2).
// lane 0-15  : row = lane,    K = kbase+0..7  and kbase+16..23
// lane 16-31 : row = lane-16, K = kbase+8..15 and kbase+24..31  (kbase=(l>>4)*8)
// B fragment uses the identical pattern on B^T (column-major B), so both
// operands are two contiguous 16-byte loads (global_load_b128).
// ---------------------------------------------------------------------------
__device__ __forceinline__ v16h load_frag_f16(const _Float16* __restrict__ base,
                                              int ld, int tile_row, int k0, int lane) {
  const _Float16* p = base + (size_t)(tile_row + (lane & 15)) * (size_t)ld
                           + (size_t)(k0 + ((lane >> 4) << 3));
  v8h lo = *reinterpret_cast<const v8h*>(p);
  v8h hi = *reinterpret_cast<const v8h*>(p + 16);
  return __builtin_shufflevector(lo, hi, 0,1,2,3,4,5,6,7,8,9,10,11,12,13,14,15);
}

__device__ __forceinline__ void store_tile_f32(float* __restrict__ out, int ld,
                                               int row0, int col0, const v8f& c,
                                               float scale, int lane) {
  int crow = row0 + ((lane >> 4) << 3);
  int ccol = col0 + (lane & 15);
  #pragma unroll
  for (int r = 0; r < 8; ++r)
    out[(size_t)(crow + r) * ld + ccol] = c[r] * scale;
}

__device__ __forceinline__ void store_tile_f16(_Float16* __restrict__ out, int ld,
                                               int row0, int col0, const v8f& c, int lane) {
  int crow = row0 + ((lane >> 4) << 3);
  int ccol = col0 + (lane & 15);
  #pragma unroll
  for (int r = 0; r < 8; ++r)
    out[(size_t)(crow + r) * ld + ccol] = (_Float16)c[r];
}

// ------------------------- point-net stage 1 -------------------------------
__global__ void k_fc1(const float* __restrict__ xyz, const float* __restrict__ W1,
                      const float* __restrict__ b1, float* __restrict__ t1) {
  int idx = blockIdx.x * blockDim.x + threadIdx.x;      // BN*HH elements
  int i = idx >> 7;
  int j = idx & (HH - 1);
  const float* x = xyz + 3 * i;
  t1[idx] = fmaf(x[0], W1[j], fmaf(x[1], W1[HH + j], fmaf(x[2], W1[2*HH + j], b1[j])));
}

// column sums / sums-of-squares for BatchNorm (training-mode, biased var)
__global__ void k_colstats(const float* __restrict__ x, int rows, int cols,
                           float* __restrict__ sum, float* __restrict__ sq) {
  int lanes = blockDim.x / cols;            // threads sharing a column
  int j  = threadIdx.x % cols;
  int rl = threadIdx.x / cols;
  int rpb = rows / gridDim.x;
  int r0 = blockIdx.x * rpb;
  float s = 0.f, q = 0.f;
  for (int r = r0 + rl; r < r0 + rpb; r += lanes) {
    float v = x[(size_t)r * cols + j];
    s += v; q += v * v;
  }
  __shared__ float ls[256], lq[256];
  ls[threadIdx.x] = s; lq[threadIdx.x] = q;
  __syncthreads();
  if (rl == 0) {
    for (int t = 1; t < lanes; ++t) { s += ls[t*cols + j]; q += lq[t*cols + j]; }
    atomicAdd(&sum[j], s);
    atomicAdd(&sq[j], q);
  }
}

__global__ void k_bnfin(const float* __restrict__ sum, const float* __restrict__ sq,
                        const float* __restrict__ g, const float* __restrict__ be,
                        float* __restrict__ sc, float* __restrict__ sh,
                        int cols, float invRows) {
  int j = threadIdx.x;
  if (j < cols) {
    float mean = sum[j] * invRows;
    float var  = sq[j] * invRows - mean * mean;
    float s = g[j] * rsqrtf(var + 1e-5f);
    sc[j] = s;
    sh[j] = be[j] - mean * s;
  }
}

// t2 = relu(bn1(t1)) @ W2 + b2  (one block per row, BN'd row staged in LDS)
__global__ void k_fc2(const float* __restrict__ t1, const float* __restrict__ sc1,
                      const float* __restrict__ sh1, const float* __restrict__ W2,
                      const float* __restrict__ b2, float* __restrict__ t2) {
  __shared__ float a[HH];
  int i = blockIdx.x;
  int j = threadIdx.x;
  if (j < HH) {
    float v = t1[(size_t)i * HH + j];
    a[j] = fmaxf(fmaf(v, sc1[j], sh1[j]), 0.f);
  }
  __syncthreads();
  float acc = b2[j];
  #pragma unroll 8
  for (int k = 0; k < HH; ++k)
    acc = fmaf(a[k], W2[(size_t)k * DD + j], acc);
  t2[(size_t)i * DD + j] = acc;
}

// f = relu(bn2(t2)) + features  -> f16
__global__ void k_bn2_add(const float* __restrict__ t2, const float* __restrict__ feat,
                          const float* __restrict__ sc, const float* __restrict__ sh,
                          _Float16* __restrict__ fh) {
  int idx = blockIdx.x * blockDim.x + threadIdx.x;      // BN*DD
  int j = idx & (DD - 1);
  float v = fmaxf(fmaf(t2[idx], sc[j], sh[j]), 0.f) + feat[idx];
  fh[idx] = (_Float16)v;
}

// Wq/Wk/Wv (in,out) f32 -> transposed (out,in) f16 for contiguous B-frag loads
__global__ void k_prep_w(const float* __restrict__ Wq, const float* __restrict__ Wk,
                         const float* __restrict__ Wv, _Float16* __restrict__ wT) {
  int idx = blockIdx.x * blockDim.x + threadIdx.x;      // 3*DD*DD
  int m = idx / (DD * DD);
  int r = idx - m * DD * DD;
  int o = r >> 8;
  int i = r & (DD - 1);
  const float* W = (m == 0) ? Wq : (m == 1) ? Wk : Wv;
  wT[idx] = (_Float16)W[(size_t)i * DD + o];
}

// ------------------- QKV projection (WMMA, 2x2 register tile) --------------
__global__ void k_qkv(const _Float16* __restrict__ fh, const _Float16* __restrict__ wT,
                      _Float16* __restrict__ qkv) {
  int lane = threadIdx.x & 31;
  int t = blockIdx.x * (blockDim.x >> 5) + (threadIdx.x >> 5);
  int mat = t >> 12;                 // 0..2  (4096 waves per matrix)
  int rem = t & 4095;
  int mt = rem >> 3, nt = rem & 7;   // 512 x 8 super-tiles of 32x32
  int m0 = mt * 32, n0 = nt * 32;
  const _Float16* wb = wT + (size_t)mat * DD * DD;
  _Float16* out = qkv + (size_t)mat * BN * DD;
  v8f c00 = {}, c01 = {}, c10 = {}, c11 = {};
  #pragma unroll
  for (int k0 = 0; k0 < DD; k0 += 32) {
    v16h a0 = load_frag_f16(fh, DD, m0,      k0, lane);
    v16h a1 = load_frag_f16(fh, DD, m0 + 16, k0, lane);
    v16h b0 = load_frag_f16(wb, DD, n0,      k0, lane);
    v16h b1 = load_frag_f16(wb, DD, n0 + 16, k0, lane);
    c00 = __builtin_amdgcn_wmma_f32_16x16x32_f16(false, a0, false, b0, (short)0, c00, false, false);
    c01 = __builtin_amdgcn_wmma_f32_16x16x32_f16(false, a0, false, b1, (short)0, c01, false, false);
    c10 = __builtin_amdgcn_wmma_f32_16x16x32_f16(false, a1, false, b0, (short)0, c10, false, false);
    c11 = __builtin_amdgcn_wmma_f32_16x16x32_f16(false, a1, false, b1, (short)0, c11, false, false);
  }
  store_tile_f16(out, DD, m0,      n0,      c00, lane);
  store_tile_f16(out, DD, m0,      n0 + 16, c01, lane);
  store_tile_f16(out, DD, m0 + 16, n0,      c10, lane);
  store_tile_f16(out, DD, m0 + 16, n0 + 16, c11, lane);
}

// V -> V^T per batch (LDS tiled transpose, coalesced both directions)
__global__ void k_vt(const _Float16* __restrict__ vh, _Float16* __restrict__ vt) {
  __shared__ _Float16 tile[32][33];
  int n0 = blockIdx.x * 32, d0 = blockIdx.y * 32, b = blockIdx.z;
  for (int yy = threadIdx.y; yy < 32; yy += 8)
    tile[yy][threadIdx.x] = vh[((size_t)b * NN + n0 + yy) * DD + d0 + threadIdx.x];
  __syncthreads();
  for (int yy = threadIdx.y; yy < 32; yy += 8)
    vt[(size_t)b * DD * NN + (size_t)(d0 + yy) * NN + n0 + threadIdx.x] = tile[threadIdx.x][yy];
}

// -------------- S = (Q K^T) * 1/sqrt(d)  (WMMA, 2x2 register tile) ---------
__global__ void k_qkt(const _Float16* __restrict__ qh, const _Float16* __restrict__ kh,
                      float* __restrict__ S, int b) {
  int lane = threadIdx.x & 31;
  int t = blockIdx.x * (blockDim.x >> 5) + (threadIdx.x >> 5);
  int mt = t >> 7, nt = t & 127;     // 128 x 128 super-tiles of 32x32
  int m0 = mt * 32, n0 = nt * 32;
  const _Float16* qb = qh + (size_t)b * NN * DD;
  const _Float16* kb = kh + (size_t)b * NN * DD;   // rows of K == cols of K^T
  v8f c00 = {}, c01 = {}, c10 = {}, c11 = {};
  #pragma unroll
  for (int k0 = 0; k0 < DD; k0 += 32) {
    v16h a0 = load_frag_f16(qb, DD, m0,      k0, lane);
    v16h a1 = load_frag_f16(qb, DD, m0 + 16, k0, lane);
    v16h b0 = load_frag_f16(kb, DD, n0,      k0, lane);
    v16h b1 = load_frag_f16(kb, DD, n0 + 16, k0, lane);
    c00 = __builtin_amdgcn_wmma_f32_16x16x32_f16(false, a0, false, b0, (short)0, c00, false, false);
    c01 = __builtin_amdgcn_wmma_f32_16x16x32_f16(false, a0, false, b1, (short)0, c01, false, false);
    c10 = __builtin_amdgcn_wmma_f32_16x16x32_f16(false, a1, false, b0, (short)0, c10, false, false);
    c11 = __builtin_amdgcn_wmma_f32_16x16x32_f16(false, a1, false, b1, (short)0, c11, false, false);
  }
  const float sc = 0.0625f;   // 1/sqrt(256)
  store_tile_f32(S, NN, m0,      n0,      c00, sc, lane);
  store_tile_f32(S, NN, m0,      n0 + 16, c01, sc, lane);
  store_tile_f32(S, NN, m0 + 16, n0,      c10, sc, lane);
  store_tile_f32(S, NN, m0 + 16, n0 + 16, c11, sc, lane);
}

// row softmax; row held in registers; P written f16 IN PLACE over the first
// half of the same S row (stride 2*NN halves) -> no extra buffer, no races.
__global__ void k_softmax(const float* __restrict__ S, _Float16* __restrict__ P) {
  int row = blockIdx.x, tid = threadIdx.x;
  const float* sr = S + (size_t)row * NN;
  _Float16* pr = P + (size_t)row * (2 * NN);
  float v[16];
  float m = -3.4e38f;
  #pragma unroll
  for (int t = 0; t < 16; ++t) { v[t] = sr[tid + (t << 8)]; m = fmaxf(m, v[t]); }
  __shared__ float red[256];
  red[tid] = m; __syncthreads();
  for (int s = 128; s > 0; s >>= 1) { if (tid < s) red[tid] = fmaxf(red[tid], red[tid + s]); __syncthreads(); }
  m = red[0]; __syncthreads();
  float sum = 0.f;
  #pragma unroll
  for (int t = 0; t < 16; ++t) { v[t] = __expf(v[t] - m); sum += v[t]; }
  red[tid] = sum; __syncthreads();
  for (int s = 128; s > 0; s >>= 1) { if (tid < s) red[tid] += red[tid + s]; __syncthreads(); }
  float inv = 1.0f / red[0];
  #pragma unroll
  for (int t = 0; t < 16; ++t) pr[tid + (t << 8)] = (_Float16)(v[t] * inv);
}

// ------------- O = P V  (WMMA, K = 4096, 2x2 register tile) ----------------
__global__ void k_pv(const _Float16* __restrict__ P, const _Float16* __restrict__ vt,
                     float* __restrict__ out, int b) {
  int lane = threadIdx.x & 31;
  int t = blockIdx.x * (blockDim.x >> 5) + (threadIdx.x >> 5);
  int mt = t >> 3, nt = t & 7;       // 128 x 8 super-tiles of 32x32
  int m0 = mt * 32, n0 = nt * 32;
  const _Float16* vb = vt + (size_t)b * DD * NN;
  v8f c00 = {}, c01 = {}, c10 = {}, c11 = {};
  #pragma unroll 2
  for (int k0 = 0; k0 < NN; k0 += 32) {
    v16h a0 = load_frag_f16(P,  2 * NN, m0,      k0, lane);  // ldP = 2*NN (in-place f16)
    v16h a1 = load_frag_f16(P,  2 * NN, m0 + 16, k0, lane);
    v16h b0 = load_frag_f16(vb, NN,     n0,      k0, lane);
    v16h b1 = load_frag_f16(vb, NN,     n0 + 16, k0, lane);
    c00 = __builtin_amdgcn_wmma_f32_16x16x32_f16(false, a0, false, b0, (short)0, c00, false, false);
    c01 = __builtin_amdgcn_wmma_f32_16x16x32_f16(false, a0, false, b1, (short)0, c01, false, false);
    c10 = __builtin_amdgcn_wmma_f32_16x16x32_f16(false, a1, false, b0, (short)0, c10, false, false);
    c11 = __builtin_amdgcn_wmma_f32_16x16x32_f16(false, a1, false, b1, (short)0, c11, false, false);
  }
  float* ob = out + (size_t)b * NN * DD;
  store_tile_f32(ob, DD, m0,      n0,      c00, 1.0f, lane);
  store_tile_f32(ob, DD, m0,      n0 + 16, c01, 1.0f, lane);
  store_tile_f32(ob, DD, m0 + 16, n0,      c10, 1.0f, lane);
  store_tile_f32(ob, DD, m0 + 16, n0 + 16, c11, 1.0f, lane);
}

// ---------------------------------------------------------------------------
extern "C" void kernel_launch(void* const* d_in, const int* in_sizes, int n_in,
                              void* d_out, int out_size, void* d_ws, size_t ws_size,
                              hipStream_t stream) {
  (void)in_sizes; (void)n_in; (void)out_size; (void)ws_size;
  const float* xyz  = (const float*)d_in[0];
  const float* feat = (const float*)d_in[1];
  const float* W1   = (const float*)d_in[2];
  const float* b1   = (const float*)d_in[3];
  const float* g1   = (const float*)d_in[4];
  const float* be1  = (const float*)d_in[5];
  const float* W2   = (const float*)d_in[6];
  const float* b2   = (const float*)d_in[7];
  const float* g2   = (const float*)d_in[8];
  const float* be2  = (const float*)d_in[9];
  const float* Wq   = (const float*)d_in[10];
  const float* Wk   = (const float*)d_in[11];
  const float* Wv   = (const float*)d_in[12];
  float* out = (float*)d_out;

  char* ws = (char*)d_ws;
  const size_t MB = 1024ull * 1024ull;
  // [0, 64MB): t1 (8MB) | t2 (+8MB, 16MB); later entirely reused as S (64MB, L2-resident)
  float* t1 = (float*)(ws);
  float* t2 = (float*)(ws + 8 * MB);
  float* S  = (float*)(ws);
  float* st = (float*)(ws + 64 * MB);                        // stats: 1536 floats
  _Float16* fh  = (_Float16*)(ws + 64 * MB + 65536);         // 8MB
  _Float16* wT  = (_Float16*)(ws + 72 * MB + 65536);         // 384KB (reserve 512KB)
  _Float16* qkv = (_Float16*)(ws + 72 * MB + 65536 + 512 * 1024); // 24MB
  _Float16* vt  = (_Float16*)(ws + 96 * MB + 65536 + 512 * 1024); // 8MB

  float* s1sum = st;        float* s1sq = st + HH;
  float* s1sc  = st + 2*HH; float* s1sh = st + 3*HH;
  float* s2sum = st + 512;        float* s2sq = st + 512 + DD;
  float* s2sc  = st + 512 + 2*DD; float* s2sh = st + 512 + 3*DD;

  hipMemsetAsync(st, 0, 1536 * sizeof(float), stream);

  k_fc1<<<(BN * HH) / 256, 256, 0, stream>>>(xyz, W1, b1, t1);
  k_colstats<<<32, 256, 0, stream>>>(t1, BN, HH, s1sum, s1sq);
  k_bnfin<<<1, 256, 0, stream>>>(s1sum, s1sq, g1, be1, s1sc, s1sh, HH, 1.0f / BN);
  k_fc2<<<BN, 256, 0, stream>>>(t1, s1sc, s1sh, W2, b2, t2);
  k_colstats<<<32, 256, 0, stream>>>(t2, BN, DD, s2sum, s2sq);
  k_bnfin<<<1, 256, 0, stream>>>(s2sum, s2sq, g2, be2, s2sc, s2sh, DD, 1.0f / BN);
  k_bn2_add<<<(BN * DD) / 256, 256, 0, stream>>>(t2, feat, s2sc, s2sh, fh);
  k_prep_w<<<(3 * DD * DD) / 256, 256, 0, stream>>>(Wq, Wk, Wv, wT);

  // 3 * 512 * 8 = 12288 waves -> 1536 blocks (8 waves/block)
  k_qkv<<<(3 * (BN / 32) * (DD / 32)) / 8, 256, 0, stream>>>(fh, wT, qkv);
  _Float16* qh = qkv;
  _Float16* kh = qkv + (size_t)BN * DD;
  _Float16* vh = qkv + 2 * (size_t)BN * DD;
  k_vt<<<dim3(NN / 32, DD / 32, BB), dim3(32, 8), 0, stream>>>(vh, vt);

  for (int b = 0; b < BB; ++b) {
    // 128*128 = 16384 waves -> 2048 blocks
    k_qkt<<<((NN / 32) * (NN / 32)) / 8, 256, 0, stream>>>(qh, kh, S, b);
    k_softmax<<<NN, 256, 0, stream>>>(S, (_Float16*)S);
    // 128*8 = 1024 waves -> 128 blocks
    k_pv<<<((NN / 32) * (DD / 32)) / 8, 256, 0, stream>>>((const _Float16*)S, vt, out, b);
  }
}
